// MiniFNO2d_64132451664277
// MI455X (gfx1250) — matrix operand order repaired and checked by reading, not compiled, not run on hardware
//
#include <hip/hip_runtime.h>

typedef __bf16 bf16_t;
typedef __bf16 v16bf __attribute__((ext_vector_type(16)));
typedef __bf16 v8bf  __attribute__((ext_vector_type(8)));
typedef float  v8f   __attribute__((ext_vector_type(8)));
typedef float  v4f   __attribute__((ext_vector_type(4)));

#define H_IMG 256
#define W_IMG 256
#define CH    64
#define BATCH 8
#define BC    (BATCH*CH)
#define MODES 16
#define KH    31          // 16 positive + 15 negative kh rows
#define KW    16
#define NPOS  (KH*KW)     // 496
#define LDK   40          // padded LDS k-stride (80B rows -> conflict-free b128 frag reads)

// ---- workspace layout (float elements) ----
static constexpr size_t OFF_FH   = 0;                                  // [64][256]
static constexpr size_t OFF_FW   = OFF_FH   + (size_t)64*256;          // [256][32]
static constexpr size_t OFF_E2   = OFF_FW   + (size_t)256*32;          // [256][64]
static constexpr size_t OFF_G    = OFF_E2   + (size_t)256*64;          // [32][256]
static constexpr size_t OFF_WCAT = OFF_G    + (size_t)32*256;          // [496][64][128]
static constexpr size_t OFF_A1   = OFF_WCAT + (size_t)NPOS*64*128;     // [512][64][256]
static constexpr size_t OFF_XCAT = OFF_A1   + (size_t)BC*64*256;       // [496][128][32]
static constexpr size_t OFF_OFP  = OFF_XCAT + (size_t)NPOS*128*32;     // [512][64][32]
static constexpr size_t OFF_Y    = OFF_OFP  + (size_t)BC*64*32;        // [512][256][32]
// total = OFF_Y + 512*256*32 floats  (~79 MB)

__device__ __forceinline__ float angf(int m) {
    // 2*pi*(m mod 256)/256
    return (float)(m & 255) * 0.024543692606170259f;
}

__device__ __forceinline__ void splitf(float f, bf16_t& hi, bf16_t& lo) {
    hi = (bf16_t)f;
    lo = (bf16_t)(f - (float)hi);
}

// A-fragment: l16 = row M, half selects K range; LDS tile row-major [rows][LDK]
__device__ __forceinline__ v16bf load_afrag(const bf16_t* sA, int row, int half) {
    v8bf a0 = *(const v8bf*)(sA + row*LDK + half*8);        // K = half*8 .. +7
    v8bf a1 = *(const v8bf*)(sA + row*LDK + 16 + half*8);   // K = 16+half*8 .. +7
    return __builtin_shufflevector(a0, a1, 0,1,2,3,4,5,6,7,8,9,10,11,12,13,14,15);
}

// B-fragment from transposed LDS tile BT[n][LDK]; l16 = col N, half selects K range
__device__ __forceinline__ v16bf load_bfrag(const bf16_t* sBT, int n, int half) {
    v8bf b0 = *(const v8bf*)(sBT + n*LDK + half*8);         // K = half*8 .. +7
    v8bf b1 = *(const v8bf*)(sBT + n*LDK + 16 + half*8);    // K = 16+half*8 .. +7
    return __builtin_shufflevector(b0, b1, 0,1,2,3,4,5,6,7,8,9,10,11,12,13,14,15);
}

// split-precision triple-WMMA: A*B ~= Ah*Bh + Ah*Bl + Al*Bh  (f32 accumulate)
__device__ __forceinline__ v8f wmma3(v16bf ah, v16bf al, v16bf bh, v16bf bl, v8f acc) {
    acc = __builtin_amdgcn_wmma_f32_16x16x32_bf16(false, ah, false, bh, (short)0, acc, false, false);
    acc = __builtin_amdgcn_wmma_f32_16x16x32_bf16(false, ah, false, bl, (short)0, acc, false, false);
    acc = __builtin_amdgcn_wmma_f32_16x16x32_bf16(false, al, false, bh, (short)0, acc, false, false);
    return acc;
}

// ===================== init kernels =====================

__global__ void fno_init_basis(float* ws) {
    const int i = blockIdx.x * blockDim.x + threadIdx.x;   // up to 16384
    const float s = 1.0f / 256.0f;                         // ortho norm (once fwd, once inv)
    // Fh [64 rows][256 h]: rows 0..30 = cos(kh*h)/256, 31..61 = -sin/256, 62..63 = 0
    if (i < 64*256) {
        int r = i >> 8, h = i & 255;
        float v = 0.0f;
        if (r < 31) {
            int khv = (r < 16) ? r : 225 + r;              // 241..255
            v = __cosf(angf(khv * h)) * s;
        } else if (r < 62) {
            int rr = r - 31;
            int khv = (rr < 16) ? rr : 225 + rr;
            v = -__sinf(angf(khv * h)) * s;
        }
        ws[OFF_FH + i] = v;
    }
    // Fw [256 w][32]: cols 0..15 = cos(kw*w), 16..31 = -sin(kw*w)
    if (i < 256*32) {
        int w = i >> 5, n = i & 31;
        float v = (n < 16) ? __cosf(angf(n * w)) : -__sinf(angf((n - 16) * w));
        ws[OFF_FW + i] = v;
    }
    // E2 [256 h][64]: cols 0..30 = cos(kh*h), 32..62 = sin(kh*h), cols 31/63 = 0 (pad kill)
    if (i < 256*64) {
        int h = i >> 6, k = i & 63;
        float v = 0.0f;
        if (k < 31) {
            int khv = (k < 16) ? k : 225 + k;
            v = __cosf(angf(khv * h));
        } else if (k >= 32 && k < 63) {
            int kk = k - 32;
            int khv = (kk < 16) ? kk : 225 + kk;
            v = __sinf(angf(khv * h));
        }
        ws[OFF_E2 + i] = v;
    }
    // G [32 k][256 w]: rows 0..15 = c_k*cos(k*w)/256, rows 16..31 = -c_k*sin(k*w)/256
    if (i < 32*256) {
        int k = i >> 8, w = i & 255;
        float v;
        if (k < 16) {
            float c = (k == 0) ? 1.0f : 2.0f;
            v = c * __cosf(angf(k * w)) * s;
        } else {
            int kk = k - 16;
            float c = (kk == 0) ? 1.0f : 2.0f;
            v = -c * __sinf(angf(kk * w)) * s;
        }
        ws[OFF_G + i] = v;
    }
}

// Pack weights [o][i][16][16][2] -> Wcat[p][o][k] with k<64 = Wr[o][i], k>=64 = Wi[o][i]
__global__ void fno_init_wcat(const float* __restrict__ weights, float* __restrict__ ws) {
    const size_t total = (size_t)NPOS * 64 * 128;
    for (size_t i = (size_t)blockIdx.x * blockDim.x + threadIdx.x; i < total;
         i += (size_t)gridDim.x * blockDim.x) {
        int k  = (int)(i & 127);
        size_t t = i >> 7;
        int o  = (int)(t & 63);
        int p  = (int)(t >> 6);
        int khp = p >> 4, kwp = p & 15;
        int wh  = (khp < 16) ? khp : khp - 15;             // negative rows reuse weight rows 1..15
        int ic  = k & 63;
        int im  = k >> 6;                                  // 0 = Re, 1 = Im
        ws[OFF_WCAT + i] =
            weights[((((size_t)o*64 + ic)*MODES + wh)*MODES + kwp)*2 + im];
    }
}

// Zero the pad rows (kh index 31 / 63) of all OFP matrices
__global__ void fno_init_ofppad(float* ws) {
    const int i = blockIdx.x * blockDim.x + threadIdx.x;   // 512*2*32 = 32768
    if (i >= BC * 2 * 32) return;
    int c = i & 31, r = (i >> 5) & 1, img = i >> 6;
    ws[OFF_OFP + (size_t)img*2048 + (r ? 63 : 31)*32 + c] = 0.0f;
}

// ===================== stage A: column DFT =====================
// per image bc: C[64,256] = Fh[64,256] * X[256,256];  block handles 64 N-cols
__global__ __launch_bounds__(128) void fno_stageA(const float* __restrict__ x,
                                                  float* __restrict__ ws) {
    __shared__ __align__(64) bf16_t sA_hi[64*LDK], sA_lo[64*LDK];
    __shared__ __align__(64) bf16_t sB_hi[64*LDK], sB_lo[64*LDK];   // BT[n=64][k]
    const int bc = blockIdx.x;
    const int n0 = blockIdx.y * 64;
    const float* Fh = ws + OFF_FH;
    const float* X  = x  + (size_t)bc * (H_IMG * W_IMG);
    float*       C  = ws + OFF_A1 + (size_t)bc * (64 * 256);
    const int tid = threadIdx.x, wave = tid >> 5, lane = tid & 31;
    const int half = lane >> 4, l16 = lane & 15;
    v8f acc[4] = {};
    for (int k0 = 0; k0 < 256; k0 += 32) {
        __syncthreads();
        for (int i = tid; i < 64*32; i += 128) {           // A: coalesced rows of Fh
            int r = i >> 5, c = i & 31;
            bf16_t h, l; splitf(Fh[r*256 + k0 + c], h, l);
            sA_hi[r*LDK + c] = h; sA_lo[r*LDK + c] = l;
        }
        for (int it = 0; it < 4; ++it) {                   // B: X [32k][64n], b128 coalesced
            int idx = tid + it*128;                        // 0..511
            int k = idx >> 4, n4 = (idx & 15) << 2;
            v4f f = *(const v4f*)(X + (size_t)(k0 + k)*256 + n0 + n4);
            for (int j = 0; j < 4; ++j) {
                bf16_t h, l; splitf(f[j], h, l);
                sB_hi[(n4 + j)*LDK + k] = h; sB_lo[(n4 + j)*LDK + k] = l;
            }
        }
        __syncthreads();
        v16bf ah = load_afrag(sA_hi, wave*16 + l16, half);
        v16bf al = load_afrag(sA_lo, wave*16 + l16, half);
        for (int nt = 0; nt < 4; ++nt) {
            v16bf bh = load_bfrag(sB_hi, nt*16 + l16, half);
            v16bf bl = load_bfrag(sB_lo, nt*16 + l16, half);
            acc[nt] = wmma3(ah, al, bh, bl, acc[nt]);
        }
    }
    for (int nt = 0; nt < 4; ++nt)
        for (int v = 0; v < 8; ++v)
            C[(size_t)(wave*16 + v + 8*half)*256 + n0 + nt*16 + l16] = acc[nt][v];
}

// ===================== stage B: row DFT + complex combine =====================
// per image bc: P[64,32] = A1[64,256] * Fw[256,32]; epilogue scatters Xcat
__global__ __launch_bounds__(128) void fno_stageB(float* __restrict__ ws) {
    __shared__ __align__(64) bf16_t sA_hi[64*LDK], sA_lo[64*LDK];
    __shared__ __align__(64) bf16_t sB_hi[32*LDK], sB_lo[32*LDK];
    __shared__ float sP[64*32];
    const int bc = blockIdx.x;
    const float* A  = ws + OFF_A1 + (size_t)bc * (64 * 256);
    const float* Fw = ws + OFF_FW;
    float*    Xcat  = ws + OFF_XCAT;
    const int tid = threadIdx.x, wave = tid >> 5, lane = tid & 31;
    const int half = lane >> 4, l16 = lane & 15;
    v8f acc[2] = {};
    for (int k0 = 0; k0 < 256; k0 += 32) {
        __syncthreads();
        for (int it = 0; it < 4; ++it) {                   // A1 panel: b128 coalesced
            int idx = tid + it*128;                        // 0..511
            int r = idx >> 3, c4 = (idx & 7) << 2;
            v4f f = *(const v4f*)(A + r*256 + k0 + c4);
            for (int j = 0; j < 4; ++j) {
                bf16_t h, l; splitf(f[j], h, l);
                sA_hi[r*LDK + c4 + j] = h; sA_lo[r*LDK + c4 + j] = l;
            }
        }
        for (int i = tid; i < 32*32; i += 128) {           // Fw panel: coalesced n
            int k = i >> 5, n = i & 31;
            bf16_t h, l; splitf(Fw[(k0 + k)*32 + n], h, l);
            sB_hi[n*LDK + k] = h; sB_lo[n*LDK + k] = l;
        }
        __syncthreads();
        v16bf ah = load_afrag(sA_hi, wave*16 + l16, half);
        v16bf al = load_afrag(sA_lo, wave*16 + l16, half);
        for (int nt = 0; nt < 2; ++nt) {
            v16bf bh = load_bfrag(sB_hi, nt*16 + l16, half);
            v16bf bl = load_bfrag(sB_lo, nt*16 + l16, half);
            acc[nt] = wmma3(ah, al, bh, bl, acc[nt]);
        }
    }
    __syncthreads();
    for (int nt = 0; nt < 2; ++nt)
        for (int v = 0; v < 8; ++v)
            sP[(wave*16 + v + 8*half)*32 + nt*16 + l16] = acc[nt][v];
    __syncthreads();
    const int b = bc >> 6, c = bc & 63;
    for (int i = tid; i < NPOS; i += 128) {
        int kh = i >> 4, kw = i & 15;
        // complex combine: X = (Ar + iAi)(cos - i sin)
        float xr = sP[kh*32 + kw]      - sP[(31 + kh)*32 + 16 + kw];
        float xi = sP[kh*32 + 16 + kw] + sP[(31 + kh)*32 + kw];
        float* Xp = Xcat + (size_t)i * (128 * 32);
        Xp[c*32 + b]             = xr;
        Xp[c*32 + 16 + b]        = xi;
        Xp[(64 + c)*32 + b]      = -xi;
        Xp[(64 + c)*32 + 16 + b] = xr;
    }
}

// ===================== stage C: per-mode 64x64 complex channel mix =====================
// per position p: Cc[64,32] = Wcat_p[64,128] * Xcat_p[128,32]; scatter into OFP
__global__ __launch_bounds__(128) void fno_stageC(float* __restrict__ ws) {
    __shared__ __align__(64) bf16_t sA_hi[64*LDK], sA_lo[64*LDK];
    __shared__ __align__(64) bf16_t sB_hi[32*LDK], sB_lo[32*LDK];
    const int p = blockIdx.x;
    const float* W = ws + OFF_WCAT + (size_t)p * (64 * 128);
    const float* X = ws + OFF_XCAT + (size_t)p * (128 * 32);
    float*     OFP = ws + OFF_OFP;
    const int tid = threadIdx.x, wave = tid >> 5, lane = tid & 31;
    const int half = lane >> 4, l16 = lane & 15;
    v8f acc[2] = {};
    for (int k0 = 0; k0 < 128; k0 += 32) {
        __syncthreads();
        for (int it = 0; it < 4; ++it) {                   // W panel: b128 coalesced
            int idx = tid + it*128;
            int r = idx >> 3, c4 = (idx & 7) << 2;
            v4f f = *(const v4f*)(W + r*128 + k0 + c4);
            for (int j = 0; j < 4; ++j) {
                bf16_t h, l; splitf(f[j], h, l);
                sA_hi[r*LDK + c4 + j] = h; sA_lo[r*LDK + c4 + j] = l;
            }
        }
        for (int i = tid; i < 32*32; i += 128) {           // Xcat panel: coalesced n
            int k = i >> 5, n = i & 31;
            bf16_t h, l; splitf(X[(k0 + k)*32 + n], h, l);
            sB_hi[n*LDK + k] = h; sB_lo[n*LDK + k] = l;
        }
        __syncthreads();
        v16bf ah = load_afrag(sA_hi, wave*16 + l16, half);
        v16bf al = load_afrag(sA_lo, wave*16 + l16, half);
        for (int nt = 0; nt < 2; ++nt) {
            v16bf bh = load_bfrag(sB_hi, nt*16 + l16, half);
            v16bf bl = load_bfrag(sB_lo, nt*16 + l16, half);
            acc[nt] = wmma3(ah, al, bh, bl, acc[nt]);
        }
    }
    const int khp = p >> 4, kwp = p & 15;
    if (l16 < 8) {
        const int b = l16;
        for (int v = 0; v < 8; ++v) {
            int o = wave*16 + v + 8*half;
            size_t base = ((size_t)(b*CH + o)) * 2048;
            float vr = acc[0][v];        // out_r, column b
            float vi = acc[1][v];        // out_i, column 16+b
            OFP[base + khp*32 + kwp]              = vr;
            OFP[base + (32 + khp)*32 + 16 + kwp]  = vr;
            OFP[base + khp*32 + 16 + kwp]         = vi;
            OFP[base + (32 + khp)*32 + kwp]       = -vi;
        }
    }
}

// ===================== stage D1: inverse kh-DFT =====================
// per (b,o): Y[256,32] = E2[256,64] * OFP_bo[64,32]
__global__ __launch_bounds__(128) void fno_stageD1(float* __restrict__ ws) {
    __shared__ __align__(64) bf16_t sA_hi[256*LDK], sA_lo[256*LDK];
    __shared__ __align__(64) bf16_t sB_hi[32*LDK],  sB_lo[32*LDK];
    const int bo = blockIdx.x;
    const float* E2  = ws + OFF_E2;
    const float* OFP = ws + OFF_OFP + (size_t)bo * 2048;
    float*       Yo  = ws + OFF_Y   + (size_t)bo * (256 * 32);
    const int tid = threadIdx.x, wave = tid >> 5, lane = tid & 31;
    const int half = lane >> 4, l16 = lane & 15;
    v8f acc[4][2] = {};
    for (int k0 = 0; k0 < 64; k0 += 32) {
        __syncthreads();
        for (int it = 0; it < 16; ++it) {                  // E2 panel: b128 coalesced
            int idx = tid + it*128;                        // 0..2047
            int r = idx >> 3, c4 = (idx & 7) << 2;
            v4f f = *(const v4f*)(E2 + r*64 + k0 + c4);
            for (int j = 0; j < 4; ++j) {
                bf16_t h, l; splitf(f[j], h, l);
                sA_hi[r*LDK + c4 + j] = h; sA_lo[r*LDK + c4 + j] = l;
            }
        }
        for (int i = tid; i < 32*32; i += 128) {           // OFP panel: coalesced n
            int k = i >> 5, n = i & 31;
            bf16_t h, l; splitf(OFP[(k0 + k)*32 + n], h, l);
            sB_hi[n*LDK + k] = h; sB_lo[n*LDK + k] = l;
        }
        __syncthreads();
        v16bf bh0 = load_bfrag(sB_hi, l16, half),      bl0 = load_bfrag(sB_lo, l16, half);
        v16bf bh1 = load_bfrag(sB_hi, 16 + l16, half), bl1 = load_bfrag(sB_lo, 16 + l16, half);
        for (int j = 0; j < 4; ++j) {
            int row = (wave + 4*j)*16 + l16;
            v16bf ah = load_afrag(sA_hi, row, half);
            v16bf al = load_afrag(sA_lo, row, half);
            acc[j][0] = wmma3(ah, al, bh0, bl0, acc[j][0]);
            acc[j][1] = wmma3(ah, al, bh1, bl1, acc[j][1]);
        }
    }
    for (int j = 0; j < 4; ++j)
        for (int nt = 0; nt < 2; ++nt)
            for (int v = 0; v < 8; ++v)
                Yo[((wave + 4*j)*16 + v + 8*half)*32 + nt*16 + l16] = acc[j][nt][v];
}

// ===================== stage D2: inverse kw-DFT (Hermitian) + bias =====================
// per (b,o): out[256,256] = Y[256,32] * G[32,256]; single K step
__global__ __launch_bounds__(128) void fno_stageD2(const float* __restrict__ ws,
                                                   const float* __restrict__ bias,
                                                   float* __restrict__ out) {
    __shared__ __align__(64) bf16_t sA_hi[256*LDK], sA_lo[256*LDK];
    __shared__ __align__(64) bf16_t sB_hi[64*LDK],  sB_lo[64*LDK];
    const int bo = blockIdx.x;
    const int n0 = blockIdx.y * 64;
    const float* Y = ws + OFF_Y + (size_t)bo * (256 * 32);
    const float* G = ws + OFF_G;
    float*       C = out + (size_t)bo * (H_IMG * W_IMG);
    const float bv = bias[bo & (CH - 1)];
    const int tid = threadIdx.x, wave = tid >> 5, lane = tid & 31;
    const int half = lane >> 4, l16 = lane & 15;
    for (int it = 0; it < 16; ++it) {                      // Y panel: b128 coalesced
        int idx = tid + it*128;                            // 0..2047
        int r = idx >> 3, c4 = (idx & 7) << 2;
        v4f f = *(const v4f*)(Y + r*32 + c4);
        for (int j = 0; j < 4; ++j) {
            bf16_t h, l; splitf(f[j], h, l);
            sA_hi[r*LDK + c4 + j] = h; sA_lo[r*LDK + c4 + j] = l;
        }
    }
    for (int it = 0; it < 4; ++it) {                       // G panel: b128 coalesced
        int idx = tid + it*128;                            // 0..511
        int k = idx >> 4, n4 = (idx & 15) << 2;
        v4f f = *(const v4f*)(G + k*256 + n0 + n4);
        for (int j = 0; j < 4; ++j) {
            bf16_t h, l; splitf(f[j], h, l);
            sB_hi[(n4 + j)*LDK + k] = h; sB_lo[(n4 + j)*LDK + k] = l;
        }
    }
    __syncthreads();
    v16bf bh[4], bl[4];
    for (int nt = 0; nt < 4; ++nt) {
        bh[nt] = load_bfrag(sB_hi, nt*16 + l16, half);
        bl[nt] = load_bfrag(sB_lo, nt*16 + l16, half);
    }
    for (int j = 0; j < 4; ++j) {
        int row = (wave + 4*j)*16 + l16;
        v16bf ah = load_afrag(sA_hi, row, half);
        v16bf al = load_afrag(sA_lo, row, half);
        int mbase = (wave + 4*j)*16 + 8*half;
        for (int nt = 0; nt < 4; ++nt) {
            v8f acc = {};
            acc = wmma3(ah, al, bh[nt], bl[nt], acc);
            for (int v = 0; v < 8; ++v)
                C[(size_t)(mbase + v)*256 + n0 + nt*16 + l16] = acc[v] + bv;
        }
    }
}

// ===================== launch =====================
extern "C" void kernel_launch(void* const* d_in, const int* in_sizes, int n_in,
                              void* d_out, int out_size, void* d_ws, size_t ws_size,
                              hipStream_t stream) {
    (void)in_sizes; (void)n_in; (void)out_size; (void)ws_size;
    const float* x       = (const float*)d_in[0];
    const float* weights = (const float*)d_in[1];
    const float* bias    = (const float*)d_in[2];
    float* out = (float*)d_out;
    float* ws  = (float*)d_ws;

    fno_init_basis<<<64, 256, 0, stream>>>(ws);
    {
        const size_t total = (size_t)NPOS * 64 * 128;
        fno_init_wcat<<<(unsigned)((total + 255) / 256), 256, 0, stream>>>(weights, ws);
    }
    fno_init_ofppad<<<128, 256, 0, stream>>>(ws);

    fno_stageA<<<dim3(BC, 4), 128, 0, stream>>>(x, ws);
    fno_stageB<<<BC, 128, 0, stream>>>(ws);
    fno_stageC<<<NPOS, 128, 0, stream>>>(ws);
    fno_stageD1<<<BC, 128, 0, stream>>>(ws);
    fno_stageD2<<<dim3(BC, 4), 128, 0, stream>>>(ws, bias, out);
}
